// LocalInference_80410377715985
// MI455X (gfx1250) — matrix operand order repaired and checked by reading, not compiled, not run on hardware
//
#include <hip/hip_runtime.h>
#include <hip/hip_bf16.h>

// ---------------------------------------------------------------------------
// ESIM "local inference" cross-attention for MI455X (gfx1250, wave32, WMMA).
//   sim      = pre @ hypo^T                       (per batch, 384x384x768)
//   P1       = softmax_h(sim   masked by hypo_mask)
//   P2       = softmax_p(sim^T masked by pre_mask)
//   attn_pre  = (P1 @ hypo) * pre_mask
//   attn_hypo = (P2 @ pre ) * hypo_mask
// All GEMMs via v_wmma_f32_16x16x32_bf16; inputs converted f32->bf16 once.
// ---------------------------------------------------------------------------

typedef __attribute__((ext_vector_type(16))) __bf16 v16bf;
typedef __attribute__((ext_vector_type(8)))  __bf16 v8bf;
typedef __attribute__((ext_vector_type(8)))  float  v8f;

#define COMB16(lo, hi) \
    __builtin_shufflevector(lo, hi, 0, 1, 2, 3, 4, 5, 6, 7, 8, 9, 10, 11, 12, 13, 14, 15)

constexpr int   kBatch   = 128;
constexpr int   kL       = 384;   // LP == LH
constexpr int   kD       = 768;
constexpr float kNegBias = 1.0e13f;

constexpr int kNT  = kL / 16;   // 24 key tiles (N) in phase A
constexpr int kKT  = kD / 32;   // 24 K steps of 32 in phase A
constexpr int kNT2 = 24;        // phase C: 48 d-tiles done as 2 halves of 24
constexpr int kKT2 = kL / 32;   // 12 K steps of 32 in phase C

// ---------------------------------------------------------------------------
// Pass 0: f32 -> bf16 conversion producing both row-major copy and transpose.
// Tiled through LDS so both global writes are coalesced.
// grid (kD/32, kL/32, B), block (32, 8)
// ---------------------------------------------------------------------------
__global__ __launch_bounds__(256)
void cvt_tr_kernel(const float* __restrict__ in,
                   __bf16* __restrict__ outN,   // [L][D] bf16
                   __bf16* __restrict__ outT)   // [D][L] bf16
{
    __shared__ __bf16 tile[32][33];
    const size_t boff = (size_t)blockIdx.z * kL * kD;
    in   += boff;
    outN += boff;
    outT += boff;
    const int d0 = blockIdx.x * 32;
    const int l0 = blockIdx.y * 32;
    #pragma unroll
    for (int r = 0; r < 32; r += 8) {
        const int l = l0 + threadIdx.y + r;
        const int d = d0 + threadIdx.x;
        const __bf16 h = (__bf16)in[(size_t)l * kD + d];
        outN[(size_t)l * kD + d] = h;
        tile[threadIdx.y + r][threadIdx.x] = h;
    }
    __syncthreads();
    #pragma unroll
    for (int r = 0; r < 32; r += 8) {
        const int d = d0 + threadIdx.y + r;
        const int l = l0 + threadIdx.x;
        outT[(size_t)d * kL + l] = tile[threadIdx.x][threadIdx.y + r];
    }
}

// ---------------------------------------------------------------------------
// Fused attention kernel. One wave owns a 16-row query tile.
//   X  : queries  [L][D] bf16 (row major)
//   Y  : keys/vals[L][D] bf16 (row major)        -> B operand of S = X Y^T
//   YT : values^T [D][L] bf16 (row major in L)   -> B operand of O = P Y
//   qmask, kmask : f32 [B][L]
//   out: f32 [L][D]
// block = 128 threads (4 waves, 64 query rows), grid = (L/64, B)
// ---------------------------------------------------------------------------
__global__ __launch_bounds__(128)
void attn_kernel(const __bf16* __restrict__ X,
                 const __bf16* __restrict__ Y,
                 const __bf16* __restrict__ YT,
                 const float* __restrict__ qmask,
                 const float* __restrict__ kmask,
                 float* __restrict__ out)
{
    __shared__ __align__(16) __bf16 Plds[4][16 * kL];   // 48 KB: per-wave P tile

    const int tid    = threadIdx.x;
    const int wave   = tid >> 5;
    const int lane   = tid & 31;
    const int lane15 = lane & 15;
    const int hl     = lane >> 4;       // half-wave select (0: lanes 0-15)
    const int b      = blockIdx.y;
    const int mblk   = blockIdx.x * 64 + wave * 16;   // first query row of tile

    const size_t boff = (size_t)b * kL * kD;
    X     += boff;
    Y     += boff;
    YT    += boff;
    out   += boff;
    qmask += (size_t)b * kL;
    kmask += (size_t)b * kL;

    // ---------------- Phase A: S(16x384) = Xtile(16x768) @ Y^T ----------------
    const __bf16* Arow  = X + (size_t)(mblk + lane15) * kD;  // A: row per lane
    const __bf16* Bbase = Y + (size_t)lane15 * kD;           // B: col per lane

    v8f acc[kNT];
    const v8f vzero = {0.f, 0.f, 0.f, 0.f, 0.f, 0.f, 0.f, 0.f};
    #pragma unroll
    for (int n = 0; n < kNT; ++n) acc[n] = vzero;

    for (int kk = 0; kk < kD; kk += 32) {
        const v8bf alo = *(const v8bf*)(Arow + kk + 8 * hl);
        const v8bf ahi = *(const v8bf*)(Arow + kk + 16 + 8 * hl);
        const v16bf afrag = COMB16(alo, ahi);
        #pragma unroll
        for (int n = 0; n < kNT; ++n) {
            const __bf16* bp = Bbase + n * 16 * kD + kk;
            const v8bf blo = *(const v8bf*)(bp + 8 * hl);
            const v8bf bhi = *(const v8bf*)(bp + 16 + 8 * hl);
            const v16bf bfrag = COMB16(blo, bhi);
            acc[n] = __builtin_amdgcn_wmma_f32_16x16x32_bf16(
                false, afrag, false, bfrag, (short)0, acc[n], false, false);
        }
    }

    // ---------------- Phase B: masked softmax along key axis ----------------
    // C/D layout: acc[n][j] holds (row = mblk + j + 8*hl, col = n*16 + lane15)
    #pragma unroll
    for (int n = 0; n < kNT; ++n) {
        const float km   = kmask[n * 16 + lane15];
        const float bias = (km - 1.0f) * kNegBias;
        #pragma unroll
        for (int j = 0; j < 8; ++j) acc[n][j] = acc[n][j] * km + bias;
    }

    float rmax[8];
    #pragma unroll
    for (int j = 0; j < 8; ++j) rmax[j] = -3.0e38f;
    #pragma unroll
    for (int n = 0; n < kNT; ++n)
        #pragma unroll
        for (int j = 0; j < 8; ++j) rmax[j] = fmaxf(rmax[j], acc[n][j]);
    #pragma unroll
    for (int j = 0; j < 8; ++j)
        #pragma unroll
        for (int s = 1; s < 16; s <<= 1)
            rmax[j] = fmaxf(rmax[j], __shfl_xor(rmax[j], s, 32));

    float rsum[8];
    #pragma unroll
    for (int j = 0; j < 8; ++j) rsum[j] = 0.f;
    #pragma unroll
    for (int n = 0; n < kNT; ++n)
        #pragma unroll
        for (int j = 0; j < 8; ++j) {
            const float e = __expf(acc[n][j] - rmax[j]);
            acc[n][j] = e;
            rsum[j] += e;
        }
    #pragma unroll
    for (int j = 0; j < 8; ++j)
        #pragma unroll
        for (int s = 1; s < 16; s <<= 1)
            rsum[j] += __shfl_xor(rsum[j], s, 32);

    // Fold the query-side output mask into P directly (row scale).
    float rscale[8];
    #pragma unroll
    for (int j = 0; j < 8; ++j)
        rscale[j] = qmask[mblk + j + 8 * hl] / rsum[j];

    // Stage P (16x384 bf16, row major) into this wave's LDS slice.
    __bf16* Pw = &Plds[wave][0];
    #pragma unroll
    for (int n = 0; n < kNT; ++n)
        #pragma unroll
        for (int j = 0; j < 8; ++j)
            Pw[(j + 8 * hl) * kL + n * 16 + lane15] =
                (__bf16)(acc[n][j] * rscale[j]);

    __syncthreads();   // LDS visibility across lanes of the wave

    // ---------------- Phase C: O(16x768) = P(16x384) @ Y(384x768) ------------
    const __bf16* ArowP = Pw + (size_t)lane15 * kL;   // A from LDS P tile

    #pragma unroll
    for (int half = 0; half < 2; ++half) {
        v8f oacc[kNT2];
        #pragma unroll
        for (int n = 0; n < kNT2; ++n) oacc[n] = vzero;

        for (int k2 = 0; k2 < kKT2; ++k2) {
            const int kk = k2 * 32;
            const v8bf alo = *(const v8bf*)(ArowP + kk + 8 * hl);
            const v8bf ahi = *(const v8bf*)(ArowP + kk + 16 + 8 * hl);
            const v16bf afrag = COMB16(alo, ahi);
            #pragma unroll
            for (int n = 0; n < kNT2; ++n) {
                const int d = (half * kNT2 + n) * 16 + lane15;  // output column
                const __bf16* bp = YT + (size_t)d * kL + kk;
                const v8bf blo = *(const v8bf*)(bp + 8 * hl);
                const v8bf bhi = *(const v8bf*)(bp + 16 + 8 * hl);
                const v16bf bfrag = COMB16(blo, bhi);
                oacc[n] = __builtin_amdgcn_wmma_f32_16x16x32_bf16(
                    false, afrag, false, bfrag, (short)0, oacc[n], false, false);
            }
        }
        #pragma unroll
        for (int n = 0; n < kNT2; ++n) {
            const int dcol = (half * kNT2 + n) * 16 + lane15;
            #pragma unroll
            for (int j = 0; j < 8; ++j)
                out[(size_t)(mblk + j + 8 * hl) * kD + dcol] = oacc[n][j];
        }
    }
}

// ---------------------------------------------------------------------------
extern "C" void kernel_launch(void* const* d_in, const int* in_sizes, int n_in,
                              void* d_out, int out_size, void* d_ws, size_t ws_size,
                              hipStream_t stream)
{
    const float* pre       = (const float*)d_in[0];
    const float* pre_mask  = (const float*)d_in[1];
    const float* hypo      = (const float*)d_in[2];
    const float* hypo_mask = (const float*)d_in[3];

    const size_t N = (size_t)kBatch * kL * kD;

    float* out_pre  = (float*)d_out;
    float* out_hypo = out_pre + N;

    // Workspace layout (bf16): preB | hypoB | preT | hypoT  (4 * 75.5 MB)
    __bf16* w     = (__bf16*)d_ws;
    __bf16* preB  = w;
    __bf16* hypoB = w + N;
    __bf16* preT  = w + 2 * N;
    __bf16* hypoT = w + 3 * N;

    const dim3 cgrid(kD / 32, kL / 32, kBatch);
    const dim3 cblk(32, 8);
    cvt_tr_kernel<<<cgrid, cblk, 0, stream>>>(pre,  preB,  preT);
    cvt_tr_kernel<<<cgrid, cblk, 0, stream>>>(hypo, hypoB, hypoT);

    const dim3 agrid(kL / 64, kBatch);
    // attn_pre  = softmax_h(pre@hypo^T) @ hypo, masked
    attn_kernel<<<agrid, 128, 0, stream>>>(preB, hypoB, hypoT,
                                           pre_mask, hypo_mask, out_pre);
    // attn_hypo = softmax_p(hypo@pre^T) @ pre, masked
    attn_kernel<<<agrid, 128, 0, stream>>>(hypoB, preB, preT,
                                           hypo_mask, pre_mask, out_hypo);
}
